// MHA2D_3934190043350
// MI455X (gfx1250) — matrix operand order repaired and checked by reading, not compiled, not run on hardware
//
#include <hip/hip_runtime.h>
#include <hip/hip_bf16.h>
#include <math.h>

// ---------------------------------------------------------------------------
// MHA2D for MI455X (gfx1250, wave32, WMMA bf16 16x16x32, fp32 accumulate)
//   B=8, C=128, H=W=32 -> N=1024, heads=8, d=16
// ---------------------------------------------------------------------------

typedef __attribute__((ext_vector_type(16))) __bf16 v16bf;
typedef __attribute__((ext_vector_type(8)))  float  v8f;

static __device__ __forceinline__ unsigned short f2bf(float f) {
  unsigned int u = __float_as_uint(f);
  u += 0x7FFFu + ((u >> 16) & 1u);           // round-to-nearest-even
  return (unsigned short)(u >> 16);
}

static __device__ __forceinline__ v16bf pack_frag(const unsigned int* w) {
  union { unsigned int u[8]; v16bf v; } x;
#pragma unroll
  for (int i = 0; i < 8; ++i) x.u[i] = w[i];
  return x.v;
}

// ---------------------------------------------------------------------------
// Kernel 1: per-channel batch-norm statistics (training-mode, biased var)
// ---------------------------------------------------------------------------
__global__ __launch_bounds__(256) void bn_stats(const float* __restrict__ x,
                                                float* __restrict__ mean,
                                                float* __restrict__ rstd) {
  __shared__ float sh[512];
  const int c = blockIdx.x, tid = threadIdx.x;
  float s = 0.f, s2 = 0.f;
  for (int b = 0; b < 8; ++b) {
    const float* p = x + ((size_t)b * 128 + c) * 1024;
    for (int i = tid; i < 1024; i += 256) { float v = p[i]; s += v; s2 += v * v; }
  }
  sh[tid] = s; sh[256 + tid] = s2;
  __syncthreads();
  for (int off = 128; off > 0; off >>= 1) {
    if (tid < off) { sh[tid] += sh[tid + off]; sh[256 + tid] += sh[256 + tid + off]; }
    __syncthreads();
  }
  if (tid == 0) {
    float m   = sh[0]   * (1.0f / 8192.0f);
    float var = sh[256] * (1.0f / 8192.0f) - m * m;
    mean[c] = m;
    rstd[c] = rsqrtf(var + 1e-5f);
  }
}

// ---------------------------------------------------------------------------
// Kernel 2: fused BN + exact GELU + q/k/v 1x1-conv GEMMs (WMMA bf16)
//   Block: 256 thr (8 waves), 64 pixels x 128 outputs, wave w owns head w.
//   Outputs bf16 [b][h][n][d]; q pre-scaled by 1/sqrt(16). Stores staged
//   through LDS so all global writes are contiguous u64s.
// ---------------------------------------------------------------------------
__global__ __launch_bounds__(256) void mha_proj(
    const float* __restrict__ x, const float* __restrict__ mean,
    const float* __restrict__ rstd, const float* __restrict__ gamma,
    const float* __restrict__ beta, const float* __restrict__ wq,
    const float* __restrict__ wk, const float* __restrict__ wv,
    unsigned short* __restrict__ qb, unsigned short* __restrict__ kbuf,
    unsigned short* __restrict__ vbuf) {
  extern __shared__ unsigned short sm[];
  unsigned short* sW = sm;                   // 3 x [128][132] bf16 weights [o][c]
  unsigned short* sA = sm + 3 * 128 * 132;   // [64][132]  bf16 activations [px][c]

  const int tid = threadIdx.x;
  const int p0  = blockIdx.x * 64;           // 64 pixels per block
  const int bI  = p0 >> 10;                  // batch
  const int n0  = p0 & 1023;                 // spatial offset within image

  const float* Ws[3] = {wq, wk, wv};
#pragma unroll
  for (int m = 0; m < 3; ++m) {
    const float* W = Ws[m];
    for (int i = tid; i < 128 * 128; i += 256) {
      int o = i >> 7, c = i & 127;
      sW[m * (128 * 132) + o * 132 + c] = f2bf(W[i]);
    }
  }
  for (int i = tid; i < 64 * 128; i += 256) {
    int c = i >> 6, px = i & 63;
    float v = x[((size_t)bI * 128 + c) * 1024 + n0 + px];
    v = (v - mean[c]) * rstd[c] * gamma[c] + beta[c];
    v = 0.5f * v * (1.0f + erff(v * 0.70710678118f));   // exact GELU
    sA[px * 132 + c] = f2bf(v);
  }
  __syncthreads();

  const int wave = tid >> 5, lane = tid & 31;
  const int hf = lane >> 4, lm = lane & 15;

  v8f acc[4][3];
#pragma unroll
  for (int rt = 0; rt < 4; ++rt)
#pragma unroll
    for (int m = 0; m < 3; ++m)
#pragma unroll
      for (int r = 0; r < 8; ++r) acc[rt][m][r] = 0.0f;

#pragma unroll
  for (int ks = 0; ks < 4; ++ks) {
    const int kb0 = ks * 32;
    // B fragments (weights, [K=c][N=o]): lane col o = wave*16+lm, K pairs/VGPR
    v16bf Bf[3];
#pragma unroll
    for (int m = 0; m < 3; ++m) {
      unsigned int bu[8];
      const int o = wave * 16 + lm;
#pragma unroll
      for (int v = 0; v < 8; ++v) {
        int kk = kb0 + (v >> 2) * 16 + hf * 8 + (v & 3) * 2;
        bu[v] = *(const unsigned int*)&sW[m * (128 * 132) + o * 132 + kk];
      }
      Bf[m] = pack_frag(bu);
    }
#pragma unroll
    for (int rt = 0; rt < 4; ++rt) {
      unsigned int au[8];
      const int px = rt * 16 + lm;
#pragma unroll
      for (int v = 0; v < 8; ++v) {
        int kk = kb0 + (v >> 2) * 16 + hf * 8 + (v & 3) * 2;
        au[v] = *(const unsigned int*)&sA[px * 132 + kk];
      }
      v16bf Af = pack_frag(au);
#pragma unroll
      for (int m = 0; m < 3; ++m)
        acc[rt][m] = __builtin_amdgcn_wmma_f32_16x16x32_bf16(
            false, Af, false, Bf[m], (short)0, acc[rt][m], false, false);
    }
  }

  // ---- store via LDS restage so global writes are contiguous u64 ----------
  unsigned short* sOut = sm;                 // reuse: [head=8][px=64][d=16]
  unsigned short* dsts[3] = {qb, kbuf, vbuf};
  const unsigned long long* src8 = (const unsigned long long*)sOut;
#pragma unroll
  for (int m = 0; m < 3; ++m) {
    __syncthreads();                         // LDS region reuse fence
    const float sc = (m == 0) ? 0.25f : 1.0f;   // q * 1/sqrt(16)
#pragma unroll
    for (int rt = 0; rt < 4; ++rt)
#pragma unroll
      for (int r = 0; r < 8; ++r) {
        int px = rt * 16 + r + 8 * hf;
        sOut[wave * 1024 + px * 16 + lm] = f2bf(acc[rt][m][r] * sc);
      }
    __syncthreads();
    unsigned long long* dst8 = (unsigned long long*)dsts[m];
    for (int i = tid; i < 2048; i += 256) {  // u64 = 4 bf16, fully coalesced
      int head = i >> 8;
      dst8[(((size_t)bI * 8 + head) * 1024 + n0) * 4 + (i & 255)] = src8[i];
    }
  }
}

// ---------------------------------------------------------------------------
// Kernel 3: flash attention + rel-pos bias + residual.
//   Block: 256 thr (8 waves) -> one (b,h, 128-query chunk). K/V staged in LDS.
//   32 keys per iteration: 2 half-K QK^T WMMAs feed 1 full-K (32) PV WMMA.
// ---------------------------------------------------------------------------
__global__ __launch_bounds__(256) void mha_attn(
    const float* __restrict__ x, const unsigned short* __restrict__ qb,
    const unsigned short* __restrict__ kbuf,
    const unsigned short* __restrict__ vbuf,
    const float* __restrict__ peh, const float* __restrict__ pew,
    float* __restrict__ out) {
  extern __shared__ unsigned char smc[];
  unsigned short* sK  = (unsigned short*)smc;   // [1024][20] bf16 (K, [n][d])
  unsigned short* sVt = sK + 1024 * 20;         // [16][1032] bf16 (V^T, [d][n])
  unsigned short* sP  = sVt + 16 * 1032;        // 8 waves x [16][36] bf16
  float* sEh = (float*)(sP + 8 * 16 * 36);      // 64 floats
  float* sEw = sEh + 64;                        // 64 floats
  float* sO  = sEw + 64;                        // [16][132] f32 output staging

  const int tid = threadIdx.x;
  const int bh = blockIdx.x >> 3;
  const int qc = blockIdx.x & 7;
  const int bI = bh >> 3, hI = bh & 7;
  const int q0 = qc * 128;

  // Prefetch the residual x lines needed at the very end (16 rows x 512B).
  if (tid < 64) {
    int cl = tid >> 2, seg = tid & 3;
    __builtin_prefetch(
        &x[(((size_t)bI * 128 + hI * 16 + cl) * 1024) + q0 + seg * 32], 0, 3);
  }

  // Stage K ([n][20], padded) and V^T ([d][1032]) with 8-byte global loads.
  const unsigned long long* Kg8 =
      (const unsigned long long*)(kbuf + (size_t)bh * 16384);
  const unsigned long long* Vg8 =
      (const unsigned long long*)(vbuf + (size_t)bh * 16384);
  for (int i = tid; i < 4096; i += 256) {
    int n = i >> 2, dq = (i & 3) * 4;
    unsigned long long kv = Kg8[i];
    *(unsigned long long*)&sK[n * 20 + dq] = kv;     // 8B-aligned (40n + 8k)
    unsigned long long vv = Vg8[i];
    const unsigned short* vp = (const unsigned short*)&vv;
#pragma unroll
    for (int j = 0; j < 4; ++j) sVt[(dq + j) * 1032 + n] = vp[j];
  }
  if (tid < 63) { sEh[tid] = peh[tid]; sEw[tid] = pew[tid]; }
  __syncthreads();

  const int wave = tid >> 5, lane = tid & 31;
  const int hf = lane >> 4, lm = lane & 15;
  const int qrow = q0 + wave * 16;

  // Resident Q A-fragment: M = query row, K = d (upper 16 K zero-padded).
  unsigned int au[8];
#pragma unroll
  for (int v = 0; v < 8; ++v) {
    int d = v * 2;
    au[v] = (hf == 0) ? *(const unsigned int*)&qb[((size_t)bh * 1024 + (qrow + lm)) * 16 + d] : 0u;
  }
  const v16bf Aq = pack_frag(au);

  int qh[8], qw[8];
#pragma unroll
  for (int r = 0; r < 8; ++r) {
    int nq = qrow + r + 8 * hf;
    qh[r] = nq >> 5; qw[r] = nq & 31;
  }

  float mrow[8], lrow[8];
  v8f Oacc;
#pragma unroll
  for (int r = 0; r < 8; ++r) { mrow[r] = -INFINITY; lrow[r] = 0.0f; Oacc[r] = 0.0f; }

  unsigned short* sPw = sP + wave * (16 * 36);

  for (int kt = 0; kt < 32; ++kt) {
    const int nb = kt * 32;

    // ---- two K B-fragments (16 keys each); K(d) rows 16..31 zero-padded ----
    unsigned int bu0[8], bu1[8];
    if (hf == 0) {
      const int n0k = nb + lm, n1k = nb + 16 + lm;
#pragma unroll
      for (int v = 0; v < 8; ++v) {
        bu0[v] = *(const unsigned int*)&sK[n0k * 20 + v * 2];
        bu1[v] = *(const unsigned int*)&sK[n1k * 20 + v * 2];
      }
    } else {
#pragma unroll
      for (int v = 0; v < 8; ++v) { bu0[v] = 0u; bu1[v] = 0u; }
    }
    v16bf Bk0 = pack_frag(bu0), Bk1 = pack_frag(bu1);

    v8f S0, S1;
#pragma unroll
    for (int r = 0; r < 8; ++r) { S0[r] = 0.0f; S1[r] = 0.0f; }
    S0 = __builtin_amdgcn_wmma_f32_16x16x32_bf16(false, Aq, false, Bk0,
                                                 (short)0, S0, false, false);
    S1 = __builtin_amdgcn_wmma_f32_16x16x32_bf16(false, Aq, false, Bk1,
                                                 (short)0, S1, false, false);

    // ---- bias + online softmax over 32 columns at once ---------------------
    const int k0 = nb + lm, k1 = nb + 16 + lm;
    const int kh0 = k0 >> 5, kw0 = k0 & 31;
    const int kh1 = k1 >> 5, kw1 = k1 & 31;
#pragma unroll
    for (int r = 0; r < 8; ++r) {
      float s0 = S0[r] + sEh[qh[r] + 31 - kh0] + sEw[qw[r] + 31 - kw0];
      float s1 = S1[r] + sEh[qh[r] + 31 - kh1] + sEw[qw[r] + 31 - kw1];
      float t = fmaxf(s0, s1);
#pragma unroll
      for (int msk = 1; msk < 16; msk <<= 1) t = fmaxf(t, __shfl_xor(t, msk, 32));
      float mnew  = fmaxf(mrow[r], t);
      float alpha = __expf(mrow[r] - mnew);
      float p0 = __expf(s0 - mnew), p1 = __expf(s1 - mnew);
      float ps = p0 + p1;
#pragma unroll
      for (int msk = 1; msk < 16; msk <<= 1) ps += __shfl_xor(ps, msk, 32);
      lrow[r] = lrow[r] * alpha + ps;
      mrow[r] = mnew;
      Oacc[r] *= alpha;
      const int row = r + 8 * hf;
      sPw[row * 36 + lm]      = f2bf(p0);        // C-layout -> LDS
      sPw[row * 36 + 16 + lm] = f2bf(p1);
    }

    // ---- P A-fragment, full K=32 (per-wave LDS is in-order) ----------------
    unsigned int pu[8];
#pragma unroll
    for (int v = 0; v < 8; ++v) {
      int kk = (v >> 2) * 16 + hf * 8 + (v & 3) * 2;
      pu[v] = *(const unsigned int*)&sPw[lm * 36 + kk];
    }
    v16bf Ap = pack_frag(pu);

    // ---- V B-fragment, full K=32: rows n = nb + (v>>2)*16 + hf*8 + (v&3)*2 -
    unsigned int vu[8];
#pragma unroll
    for (int v = 0; v < 8; ++v)
      vu[v] = *(const unsigned int*)&sVt[lm * 1032 + nb + (v >> 2) * 16 + hf * 8 + (v & 3) * 2];
    v16bf Bv = pack_frag(vu);

    Oacc = __builtin_amdgcn_wmma_f32_16x16x32_bf16(false, Ap, false, Bv,
                                                   (short)0, Oacc, false, false);
  }

  // ---- finalize into LDS, then fully coalesced residual-add writeback ------
#pragma unroll
  for (int r = 0; r < 8; ++r)
    sO[lm * 132 + wave * 16 + 8 * hf + r] = Oacc[r] * (1.0f / lrow[r]);
  __syncthreads();

  for (int i = tid; i < 2048; i += 256) {
    int cl = i >> 7, q = i & 127;
    size_t idx = ((size_t)bI * 128 + hI * 16 + cl) * 1024 + q0 + q;
    out[idx] = x[idx] + sO[cl * 132 + q];
  }
}

// ---------------------------------------------------------------------------
extern "C" void kernel_launch(void* const* d_in, const int* in_sizes, int n_in,
                              void* d_out, int out_size, void* d_ws,
                              size_t ws_size, hipStream_t stream) {
  (void)in_sizes; (void)n_in; (void)out_size; (void)ws_size;
  const float* x     = (const float*)d_in[0];
  const float* gamma = (const float*)d_in[1];
  const float* beta  = (const float*)d_in[2];
  const float* wq    = (const float*)d_in[3];
  const float* wk    = (const float*)d_in[4];
  const float* wv    = (const float*)d_in[5];
  const float* peh   = (const float*)d_in[6];
  const float* pew   = (const float*)d_in[7];
  float* out = (float*)d_out;

  float* wsf  = (float*)d_ws;
  float* mean = wsf;
  float* rstd = wsf + 128;
  unsigned short* qb = (unsigned short*)(wsf + 256);
  const size_t qkv_elems = (size_t)8 * 8 * 1024 * 16;   // 1,048,576 bf16 each
  unsigned short* kbuf = qb + qkv_elems;
  unsigned short* vbuf = kbuf + qkv_elems;

  bn_stats<<<128, 256, 0, stream>>>(x, mean, rstd);

  const size_t smemB = (size_t)(3 * 128 * 132 + 64 * 132) * sizeof(unsigned short);
  mha_proj<<<128, 256, smemB, stream>>>(x, mean, rstd, gamma, beta, wq, wk, wv,
                                        qb, kbuf, vbuf);

  const size_t smemC =
      (size_t)(1024 * 20 + 16 * 1032 + 8 * 16 * 36) * sizeof(unsigned short) +
      (size_t)(64 + 64 + 16 * 132) * sizeof(float);
  mha_attn<<<512, 256, smemC, stream>>>(x, qb, kbuf, vbuf, peh, pew, out);
}
